// LSTM_59261958750977
// MI455X (gfx1250) — matrix operand order repaired and checked by reading, not compiled, not run on hardware
//
#include <hip/hip_runtime.h>
#include <math.h>

// ---------------- constants ----------------
#define BB 256      // batch
#define TT 1024     // sequence length
#define CC 64       // timestep chunk for xw precompute / scan
#define NCHUNK (TT / CC)

typedef __attribute__((ext_vector_type(16))) __bf16 v16bf;
typedef __attribute__((ext_vector_type(8)))  __bf16 v8bf;
typedef __attribute__((ext_vector_type(8)))  float  v8f;

__device__ __forceinline__ float sigm(float x) { return 1.0f / (1.0f + __expf(-x)); }

// ---------------------------------------------------------------------------
// Conv1d (in=1, out=64, k=5, left pad 4) + ReLU, write (B,T,64) bf16 rows
// ---------------------------------------------------------------------------
__global__ __launch_bounds__(256)
void conv_relu_kernel(const float* __restrict__ x, const float* __restrict__ w,
                      const float* __restrict__ bias, __bf16* __restrict__ y)
{
    size_t idx = (size_t)blockIdx.x * blockDim.x + threadIdx.x;   // (b*T+t)*64 + d
    if (idx >= (size_t)BB * TT * 64) return;
    int d = (int)(idx & 63);
    size_t bt = idx >> 6;
    int t = (int)(bt % TT);
    int b = (int)(bt / TT);
    const float* xr = x + (size_t)b * TT;
    float s = bias[d];
#pragma unroll
    for (int k = 0; k < 5; ++k) {
        int ti = t + k - 4;
        if (ti >= 0) s += w[d * 5 + k] * xr[ti];
    }
    y[idx] = (__bf16)fmaxf(s, 0.0f);
}

// ---------------------------------------------------------------------------
// Pack an f32 weight matrix (N rows x K cols, row-major; used as B = W^T, KxN)
// into WMMA bf16 B-fragment tiles:
//   dst[((kt*ntiles + nt)*32 + lane)*16 + e] = W[n, k]
//   with n = nt*16 + lane%16 ,  k = kt*32 + (lane/16)*8 + (e%8) + (e/8)*16
// ---------------------------------------------------------------------------
__global__ __launch_bounds__(256)
void pack_b_kernel(const float* __restrict__ src, __bf16* __restrict__ dst, int N, int K)
{
    int idx = blockIdx.x * blockDim.x + threadIdx.x;
    if (idx >= N * K) return;
    int e    = idx & 15;
    int L    = (idx >> 4) & 31;
    int tile = idx >> 9;
    int ntiles = N >> 4;
    int nt = tile % ntiles;
    int kt = tile / ntiles;
    int n = nt * 16 + (L & 15);
    int k = kt * 32 + ((L >> 4) << 3) + (e & 7) + ((e >> 3) << 4);
    dst[idx] = (__bf16)src[(size_t)n * K + k];
}

// ---------------------------------------------------------------------------
// Fully parallel input-projection GEMM for one timestep chunk, both directions:
//   xw[dir][i][b][0:4H] = x[b, t(dir,i), :] @ W_ih[dir]^T + bias[dir]
// with t = c0+i (fwd) or T-1-c0-i (bwd). grid = 2 * CC * (B/16) blocks.
// Runs at full-GPU parallelism so this work stays off the serial scan path.
// ---------------------------------------------------------------------------
__global__ __launch_bounds__(256)
void xw_gemm_kernel(const __bf16* __restrict__ xin, int Din,
                    const __bf16* __restrict__ Pih,   // packed, per-dir stride 4H*Din
                    const float*  __restrict__ bias2, // (2,4H)
                    float* __restrict__ xw,           // [2][CC][B][4H]
                    int H, int c0)
{
    const int G      = 4 * H;
    const int ntiles = G >> 4;
    const int ktiles = Din >> 5;
    const int ntpw   = ntiles >> 3;

    int blk = blockIdx.x;
    const int mtile = blk % (BB / 16); blk /= (BB / 16);
    const int i   = blk % CC;
    const int dir = blk / CC;
    const int t   = dir ? (TT - 1 - c0 - i) : (c0 + i);

    const int tid  = threadIdx.x;
    const int wave = tid >> 5;
    const int lane = tid & 31;
    const int mrow = lane & 15;
    const int khalf = (lane >> 4) << 3;

    const int batch = mtile * 16 + mrow;
    const __bf16* xrow = xin + ((size_t)batch * TT + t) * Din;
    const __bf16* PihD = Pih + (size_t)dir * (size_t)G * (size_t)Din;
    const float*  bias = bias2 + dir * G;
    float* orow = xw + ((size_t)(dir * CC + i) * BB) * G;

    for (int q = 0; q < ntpw; ++q) {
        const int nt = wave * ntpw + q;
        v8f acc = {0.f, 0.f, 0.f, 0.f, 0.f, 0.f, 0.f, 0.f};
        for (int kt = 0; kt < ktiles; ++kt) {
            v8bf lo = *(const v8bf*)(xrow + kt * 32 + khalf);
            v8bf hi = *(const v8bf*)(xrow + kt * 32 + khalf + 16);
            v16bf a = __builtin_shufflevector(lo, hi, 0,1,2,3,4,5,6,7,8,9,10,11,12,13,14,15);
            v16bf b = *(const v16bf*)(PihD + (((size_t)kt * ntiles + nt) * 32 + lane) * 16);
            acc = __builtin_amdgcn_wmma_f32_16x16x32_bf16(false, a, false, b,
                                                          (short)0, acc, false, false);
        }
        const int n = nt * 16 + (lane & 15);
        const float bn = bias[n];
        const int mbase = (lane >> 4) << 3;
#pragma unroll
        for (int r = 0; r < 8; ++r) {
            orow[(size_t)(mtile * 16 + mbase + r) * G + n] = acc[r] + bn;
        }
    }
}

// ---------------------------------------------------------------------------
// Persistent recurrent scan over one chunk of CC steps. grid = 2*(B/16) WGs,
// blockIdx = mtile*2 + dir. Gates accumulator is seeded from precomputed xw
// (already includes bias), then only h@Whh^T runs via WMMA on the serial path.
// h (bf16) / c (f32) live in LDS during the chunk and persist in d_ws between
// chunk launches.
// ---------------------------------------------------------------------------
__global__ __launch_bounds__(256)
void lstm_scan_chunk(const float* __restrict__ xw,   // [2][CC][B][4H]
                     const __bf16* __restrict__ Phh, // packed, per-dir stride 4H*H
                     float* __restrict__ cstate,     // [2][B][H]
                     __bf16* __restrict__ hstate,    // [2][B][H]
                     __bf16* __restrict__ out,       // (B,T,2H)
                     int H, int hshift, int c0, int first)
{
    const int G        = 4 * H;
    const int ntiles   = G >> 4;
    const int ktiles_h = H >> 5;
    const int ntpw     = ntiles >> 3;

    const int dir   = blockIdx.x & 1;
    const int mtile = blockIdx.x >> 1;
    const int tid   = threadIdx.x;
    const int wave  = tid >> 5;
    const int lane  = tid & 31;
    const int mrow  = lane & 15;
    const int khalf = (lane >> 4) << 3;

    __shared__ float gates[16 * 512];
    __shared__ float cst[16 * 128];
    __shared__ __align__(16) __bf16 hst[16 * 128];

    // restore (or zero) persistent state
    for (int idx = tid; idx < 16 * H; idx += 256) {
        const int m = idx >> hshift;
        const int j = idx & (H - 1);
        const size_t g = (size_t)dir * BB * H + (size_t)(mtile * 16 + m) * H + j;
        if (first) { cst[idx] = 0.0f; hst[idx] = (__bf16)0.0f; }
        else       { cst[idx] = cstate[g]; hst[idx] = hstate[g]; }
    }
    __syncthreads();

    const __bf16* PhhD = Phh + (size_t)dir * (size_t)G * (size_t)H;

    for (int s = 0; s < CC; ++s) {
        const int t = dir ? (TT - 1 - c0 - s) : (c0 + s);
        const float* xwrow = xw + ((size_t)(dir * CC + s) * BB + mtile * 16) * G;

        // prefetch next step's xw rows into cache while we compute this step
        if (s + 1 < CC) {
            const float* xwnext = xw + ((size_t)(dir * CC + s + 1) * BB + mtile * 16) * G;
            __builtin_prefetch(xwnext + (size_t)tid * 32, 0, 0);
        }

        for (int q = 0; q < ntpw; ++q) {
            const int nt = wave * ntpw + q;
            const int n  = nt * 16 + (lane & 15);
            const int mbase = (lane >> 4) << 3;

            // seed accumulator fragment from precomputed xw (bias included)
            v8f acc;
#pragma unroll
            for (int r = 0; r < 8; ++r) {
                acc[r] = xwrow[(size_t)(mbase + r) * G + n];
            }
            // recurrent contribution: h_{t-1} @ Whh^T
            for (int kt = 0; kt < ktiles_h; ++kt) {
                const __bf16* hrow = hst + mrow * H + kt * 32 + khalf;
                v8bf lo = *(const v8bf*)(hrow);
                v8bf hi = *(const v8bf*)(hrow + 16);
                v16bf a = __builtin_shufflevector(lo, hi, 0,1,2,3,4,5,6,7,8,9,10,11,12,13,14,15);
                v16bf b = *(const v16bf*)(PhhD + (((size_t)kt * ntiles + nt) * 32 + lane) * 16);
                acc = __builtin_amdgcn_wmma_f32_16x16x32_bf16(false, a, false, b,
                                                              (short)0, acc, false, false);
            }
#pragma unroll
            for (int r = 0; r < 8; ++r) {
                gates[(mbase + r) * G + n] = acc[r];
            }
        }
        __syncthreads();

        // LSTM cell update + emit h
        for (int idx = tid; idx < 16 * H; idx += 256) {
            const int m = idx >> hshift;
            const int j = idx & (H - 1);
            const float gi = gates[m * G + j];
            const float gf = gates[m * G + H + j];
            const float gg = gates[m * G + 2 * H + j];
            const float go = gates[m * G + 3 * H + j];
            const float cn = sigm(gf) * cst[idx] + sigm(gi) * tanhf(gg);
            const float hn = sigm(go) * tanhf(cn);
            cst[idx] = cn;
            hst[idx] = (__bf16)hn;
            out[((size_t)(mtile * 16 + m) * TT + t) * (2 * H) + dir * H + j] = (__bf16)hn;
        }
        __syncthreads();
    }

    // persist state for next chunk
    for (int idx = tid; idx < 16 * H; idx += 256) {
        const int m = idx >> hshift;
        const int j = idx & (H - 1);
        const size_t g = (size_t)dir * BB * H + (size_t)(mtile * 16 + m) * H + j;
        cstate[g] = cst[idx];
        hstate[g] = hst[idx];
    }
}

// ---------------------------------------------------------------------------
// FC head: h = y3[:, -1, :]  (256 x 128)  -> 512 relu -> 256 relu -> 128 relu -> 2
// ---------------------------------------------------------------------------
__global__ __launch_bounds__(128)
void head_kernel(const __bf16* __restrict__ y3,
                 const float* __restrict__ w1, const float* __restrict__ b1,
                 const float* __restrict__ w2, const float* __restrict__ b2,
                 const float* __restrict__ w3, const float* __restrict__ b3,
                 const float* __restrict__ w4, const float* __restrict__ b4,
                 float* __restrict__ out)
{
    __shared__ float a0[128], a1[512], a2[256], a3[128];
    const int b = blockIdx.x, tid = threadIdx.x;
    const __bf16* row = y3 + ((size_t)b * TT + (TT - 1)) * 128;
    if (tid < 128) a0[tid] = (float)row[tid];
    __syncthreads();
    for (int n = tid; n < 512; n += 128) {
        float s = b1[n];
        for (int j = 0; j < 128; ++j) s += w1[n * 128 + j] * a0[j];
        a1[n] = fmaxf(s, 0.0f);
    }
    __syncthreads();
    for (int n = tid; n < 256; n += 128) {
        float s = b2[n];
        for (int j = 0; j < 512; ++j) s += w2[n * 512 + j] * a1[j];
        a2[n] = fmaxf(s, 0.0f);
    }
    __syncthreads();
    for (int n = tid; n < 128; n += 128) {
        float s = b3[n];
        for (int j = 0; j < 256; ++j) s += w3[n * 256 + j] * a2[j];
        a3[n] = fmaxf(s, 0.0f);
    }
    __syncthreads();
    if (tid < 2) {
        float s = b4[tid];
        for (int j = 0; j < 128; ++j) s += w4[tid * 128 + j] * a3[j];
        out[b * 2 + tid] = s;
    }
}

// ---------------------------------------------------------------------------
extern "C" void kernel_launch(void* const* d_in, const int* in_sizes, int n_in,
                              void* d_out, int out_size, void* d_ws, size_t ws_size,
                              hipStream_t stream)
{
    (void)in_sizes; (void)n_in; (void)out_size; (void)ws_size;
    const float* x      = (const float*)d_in[0];
    const float* conv_w = (const float*)d_in[1];
    const float* conv_b = (const float*)d_in[2];
    const float* w_ih[3] = { (const float*)d_in[3], (const float*)d_in[6], (const float*)d_in[9]  };
    const float* w_hh[3] = { (const float*)d_in[4], (const float*)d_in[7], (const float*)d_in[10] };
    const float* bs[3]   = { (const float*)d_in[5], (const float*)d_in[8], (const float*)d_in[11] };
    const float* fcw[4]  = { (const float*)d_in[12], (const float*)d_in[14],
                             (const float*)d_in[16], (const float*)d_in[18] };
    const float* fcb[4]  = { (const float*)d_in[13], (const float*)d_in[15],
                             (const float*)d_in[17], (const float*)d_in[19] };

    const int Dl[3] = { 64, 256, 256 };
    const int Hl[3] = { 128, 128, 64 };
    const int Hs[3] = { 7, 7, 6 };      // log2(H)

    // ------- workspace layout (bytes) -------
    char* ws = (char*)d_ws;
    size_t off = 0;
    __bf16* buf0 = (__bf16*)(ws + off); off += (size_t)BB * TT * 64  * 2;   // conv out
    __bf16* buf1 = (__bf16*)(ws + off); off += (size_t)BB * TT * 256 * 2;   // layer1 out
    __bf16* buf2 = (__bf16*)(ws + off); off += (size_t)BB * TT * 256 * 2;   // layer2 out
    __bf16* buf3 = (__bf16*)(ws + off); off += (size_t)BB * TT * 128 * 2;   // layer3 out
    float*  xwbuf = (float*)(ws + off); off += (size_t)2 * CC * BB * 512 * 4; // xw ring
    float*  cstate = (float*)(ws + off);  off += (size_t)2 * BB * 128 * 4;
    __bf16* hstate = (__bf16*)(ws + off); off += (size_t)2 * BB * 128 * 2;
    __bf16* pk_ih[3];
    __bf16* pk_hh[3];
    for (int l = 0; l < 3; ++l) {
        const int G = 4 * Hl[l];
        pk_ih[l] = (__bf16*)(ws + off); off += (size_t)2 * G * Dl[l] * 2;
        pk_hh[l] = (__bf16*)(ws + off); off += (size_t)2 * G * Hl[l] * 2;
    }

    // ------- pack weights into WMMA B-fragment layout (bf16) -------
    for (int l = 0; l < 3; ++l) {
        const int G = 4 * Hl[l];
        for (int d = 0; d < 2; ++d) {
            {
                int NE = G * Dl[l];
                pack_b_kernel<<<(NE + 255) / 256, 256, 0, stream>>>(
                    w_ih[l] + (size_t)d * NE, pk_ih[l] + (size_t)d * NE, G, Dl[l]);
            }
            {
                int NE = G * Hl[l];
                pack_b_kernel<<<(NE + 255) / 256, 256, 0, stream>>>(
                    w_hh[l] + (size_t)d * NE, pk_hh[l] + (size_t)d * NE, G, Hl[l]);
            }
        }
    }

    // ------- conv front-end -------
    {
        size_t total = (size_t)BB * TT * 64;
        conv_relu_kernel<<<(unsigned)((total + 255) / 256), 256, 0, stream>>>(
            x, conv_w, conv_b, buf0);
    }

    // ------- three BiLSTM layers: chunked xw precompute + recurrent scan -------
    const __bf16* lin[3] = { buf0, buf1, buf2 };
    __bf16*       lout[3] = { buf1, buf2, buf3 };
    for (int l = 0; l < 3; ++l) {
        for (int c = 0; c < NCHUNK; ++c) {
            const int c0 = c * CC;
            xw_gemm_kernel<<<2 * CC * (BB / 16), 256, 0, stream>>>(
                lin[l], Dl[l], pk_ih[l], bs[l], xwbuf, Hl[l], c0);
            lstm_scan_chunk<<<2 * (BB / 16), 256, 0, stream>>>(
                xwbuf, pk_hh[l], cstate, hstate, lout[l], Hl[l], Hs[l], c0, (c == 0) ? 1 : 0);
        }
    }

    // ------- FC head -------
    head_kernel<<<BB, 128, 0, stream>>>(buf3,
        fcw[0], fcb[0], fcw[1], fcb[1], fcw[2], fcb[2], fcw[3], fcb[3],
        (float*)d_out);
}
